// decoder_74577812128044
// MI455X (gfx1250) — matrix operand order repaired and checked by reading, not compiled, not run on hardware
//
#include <hip/hip_runtime.h>
#include <hip/hip_bf16.h>
#include <cstdint>
#include <cstddef>

// ---------------- problem constants ----------------
#define H     200
#define POOL  16
#define NOUT  3200      // POOL*H
#define K1    416       // padded 400 (U part of fc2)   = 13 * 32
#define K2    224       // padded 200 (fc3 input)       = 7  * 32
#define K4    448       // [m1(200)+pad24 | m2(200)+pad24] = 14 * 32
#define NJ    13        // ceil(200/16) output column tiles
#define MT    4         // row-tiles (16 rows each) per block -> 64 rows/block

typedef unsigned short u16;
typedef __attribute__((ext_vector_type(16))) __bf16        v16bf;
typedef __attribute__((ext_vector_type(8)))  float         v8f;
typedef __attribute__((ext_vector_type(4)))  unsigned int  u32x4;

union Frag { u32x4 q[2]; v16bf v; };

__device__ __forceinline__ u16 f2bf(float f) {
  union { float f; unsigned int u; } x; x.f = f;
  unsigned int u = x.u;
  u += 0x7fffu + ((u >> 16) & 1u);   // round-to-nearest-even
  return (u16)(u >> 16);
}
__device__ __forceinline__ float sigmoidf_(float x) { return 1.0f / (1.0f + expf(-x)); }

// ---------------- preprocessing: fp32 -> bf16 padded weight/activation layouts ----
__global__ void convW(const float* __restrict__ in, int inStride, int cin,
                      u16* __restrict__ out, int cout, long long n) {
  long long i = (long long)blockIdx.x * blockDim.x + threadIdx.x;
  if (i >= n) return;
  int r = (int)(i / cout), c = (int)(i % cout);
  out[i] = (c < cin) ? f2bf(in[(long long)r * inStride + c]) : (u16)0;
}

// w4 [16,400] -> bf16 [16,448]: cols 0..199 = w4[:, :200], cols 224..423 = w4[:, 200:400]
__global__ void convW4(const float* __restrict__ in, u16* __restrict__ out) {
  int i = blockIdx.x * blockDim.x + threadIdx.x;
  if (i >= 16 * K4) return;
  int p = i / K4, c = i % K4;
  u16 r = 0;
  if (c < H)                       r = f2bf(in[p * 400 + c]);
  else if (c >= K2 && c < K2 + H)  r = f2bf(in[p * 400 + H + (c - K2)]);
  out[i] = r;
}

// scal: [0]=start [1]=end [2]=ns [3]=ne [4]=done
__global__ void init_state(float* __restrict__ h, float* __restrict__ c,
                           int* __restrict__ scal) {
  int i = threadIdx.x;
  if (i < H) { h[i] = 0.f; c[i] = 0.f; }
  if (i == 0) { scal[0] = 0; scal[1] = 0; scal[2] = 0; scal[3] = 0; scal[4] = 0; }
}

// ---------------- tiny sequential pieces (negligible FLOPs) ----------------
__global__ __launch_bounds__(800)
void lstm_step_k(const float* __restrict__ U, const float* __restrict__ Wih,
                 const float* __restrict__ Whh, const float* __restrict__ bih,
                 const float* __restrict__ bhh, const int* __restrict__ scal,
                 const float* __restrict__ h, const float* __restrict__ c,
                 float* __restrict__ ch, float* __restrict__ cc) {
  __shared__ float g[4 * H];
  int j = threadIdx.x;                       // 0..799 : one gate each
  int s = scal[0], e = scal[1];
  const float* Us = U + (long long)s * 400;
  const float* Ue = U + (long long)e * 400;
  const float* wr = Wih + (long long)j * 800;
  float acc = bih[j] + bhh[j];
  for (int k = 0; k < 400; ++k) acc += Us[k] * wr[k];
  for (int k = 0; k < 400; ++k) acc += Ue[k] * wr[400 + k];
  const float* wh = Whh + (long long)j * H;
  for (int k = 0; k < H; ++k) acc += h[k] * wh[k];
  g[j] = acc;
  __syncthreads();
  if (j < H) {
    float i_ = sigmoidf_(g[j]);
    float f_ = sigmoidf_(g[H + j]);
    float g_ = tanhf(g[2 * H + j]);
    float o_ = sigmoidf_(g[3 * H + j]);
    float cn = f_ * c[j] + i_ * g_;
    cc[j] = cn;
    ch[j] = o_ * tanhf(cn);
  }
}

// r = tanh(concat(h, U[s], U[e]) @ w1.T)   [200]
__global__ __launch_bounds__(256)
void hmn_r(const float* __restrict__ w1, const float* __restrict__ hvec,
           const float* __restrict__ U, const int* __restrict__ scal,
           float* __restrict__ r) {
  int j = threadIdx.x;
  if (j >= H) return;
  int s = scal[0], e = scal[1];
  const float* Us = U + (long long)s * 400;
  const float* Ue = U + (long long)e * 400;
  const float* w = w1 + (long long)j * 1000;
  float acc = 0.f;
  for (int k = 0; k < H; ++k)   acc += hvec[k] * w[k];
  for (int k = 0; k < 400; ++k) acc += Us[k] * w[H + k];
  for (int k = 0; k < 400; ++k) acc += Ue[k] * w[600 + k];
  r[j] = tanhf(acc);
}

// rank-factored r-part of fc2: out[j] = b2[j] + sum_k r[k] * w2[j, 400+k]   [3200]
__global__ __launch_bounds__(256)
void rw2_k(const float* __restrict__ r, const float* __restrict__ w2,
           const float* __restrict__ b2, float* __restrict__ out) {
  int j = blockIdx.x * blockDim.x + threadIdx.x;
  if (j >= NOUT) return;
  const float* w = w2 + (long long)j * 600 + 400;
  float acc = b2[j];
  for (int k = 0; k < H; ++k) acc += r[k] * w[k];
  out[j] = acc;
}

// ---------------- main WMMA kernel: GEMM + fused 16-way max-pool ----------------
// out[t, j] = max_{p<16} ( sum_k A[t,k] * B[p*200+j, k] + add[p*200+j] ),  j in [0,200)
// Block covers MT*16 = 64 rows of t (A staged once to LDS via ASYNC loads);
// 13 waves each own a 16-wide output-column tile; each B fragment feeds MT WMMAs.
__global__ __launch_bounds__(NJ * 32)
void gemm_pool(const u16* __restrict__ A, int ldA,
               const u16* __restrict__ B, const float* __restrict__ add,
               u16* __restrict__ Out, int ldOut, int outOff,
               int kSteps, int T) {
  __shared__ __align__(16) u16 sA[MT * 16 * K1];   // 64 x kTot tile, max 52 KB
  const int kTot = kSteps * 32;
  const int t0 = blockIdx.x * (MT * 16);

  // stage the 64 x kTot A tile into LDS with gfx1250 async global->LDS copies
  {
    const int kc = kTot >> 3;                      // 16B chunks per row
    for (int i = threadIdx.x; i < MT * 16 * kc; i += blockDim.x) {
      int row = i / kc, col = (i - row * kc) * 8;
      int tt = t0 + row; if (tt >= T) tt = T - 1;
      const u16* src = A + (long long)tt * ldA + col;
      unsigned dstLds = (unsigned)(size_t)(sA + row * kTot + col); // low 32b = LDS offset
      asm volatile("global_load_async_to_lds_b128 %0, %1, off"
                   :: "v"(dstLds), "v"(src) : "memory");
    }
    asm volatile("s_wait_asynccnt 0" ::: "memory");
  }
  __syncthreads();

  const int lane   = threadIdx.x & 31;
  const int wave   = threadIdx.x >> 5;
  const int laneLo = lane & 15;
  const int laneHi = lane >> 4;
  const int j0     = wave * 16;                    // this wave's output-column tile

  v8f best[MT];
#pragma unroll
  for (int mt = 0; mt < MT; ++mt)
#pragma unroll
    for (int rI = 0; rI < 8; ++rI) best[mt][rI] = -INFINITY;

  for (int p = 0; p < POOL; ++p) {
    int brow = p * H + j0 + laneLo;
    if (brow > NOUT - 1) brow = NOUT - 1;          // j>=200 lanes masked at store
    // B frag (16-bit B 32x16 layout): lane holds col N, K chunks {laneHi*8 .. +7} and {laneHi*8+16 .. +23}
    const u16* bBase = B + (long long)brow * kTot + laneHi * 8;
    if (p + 1 < POOL) {                            // hint: pull next pool's B rows
      int brow2 = (p + 1) * H + j0 + laneLo;
      if (brow2 > NOUT - 1) brow2 = NOUT - 1;
      __builtin_prefetch(B + (long long)brow2 * kTot + laneHi * 16, 0, 0);
    }

    v8f acc[MT];
#pragma unroll
    for (int mt = 0; mt < MT; ++mt)
      acc[mt] = (v8f){0.f, 0.f, 0.f, 0.f, 0.f, 0.f, 0.f, 0.f};

    for (int kk = 0; kk < kSteps; ++kk) {
      Frag fb;
      const u16* bp = bBase + kk * 32;
      fb.q[0] = *(const u32x4*)(bp);
      fb.q[1] = *(const u32x4*)(bp + 16);
#pragma unroll
      for (int mt = 0; mt < MT; ++mt) {
        // A frag (16-bit A 16x32 layout): lane holds row M=laneLo,
        // K chunks {laneHi*8 .. +7} and {laneHi*8+16 .. +23}
        Frag fa;
        const u16* ap = sA + (mt * 16 + laneLo) * kTot + laneHi * 8 + kk * 32;
        fa.q[0] = *(const u32x4*)(ap);
        fa.q[1] = *(const u32x4*)(ap + 16);
        acc[mt] = __builtin_amdgcn_wmma_f32_16x16x32_bf16(false, fa.v, false, fb.v,
                                                          (short)0, acc[mt], false, false);
      }
    }
    float av = add[brow];
#pragma unroll
    for (int mt = 0; mt < MT; ++mt)
#pragma unroll
      for (int rI = 0; rI < 8; ++rI)
        best[mt][rI] = fmaxf(best[mt][rI], acc[mt][rI] + av);
  }

  // D layout: lane holds col N=laneLo, rows M = rI + laneHi*8
  int j = j0 + laneLo;
  if (j < H) {
#pragma unroll
    for (int mt = 0; mt < MT; ++mt)
#pragma unroll
      for (int rI = 0; rI < 8; ++rI) {
        int t = t0 + mt * 16 + rI + laneHi * 8;
        if (t < T) Out[(long long)t * ldOut + outOff + j] = f2bf(best[mt][rI]);
      }
  }
}

// fc4: a[t] = max_{p<16} ( [m1|m2][t,:] . w4[p,:] + b4[p] )  -- one WMMA tile per 16 rows
__global__ __launch_bounds__(32)
void fc4_max(const u16* __restrict__ mm, const u16* __restrict__ w4,
             const float* __restrict__ b4, float* __restrict__ a, int T) {
  const int t0 = blockIdx.x * 16;
  const int lane = threadIdx.x & 31;
  const int laneLo = lane & 15;
  const int laneHi = lane >> 4;
  int tt = t0 + laneLo; if (tt >= T) tt = T - 1;
  const u16* aBase = mm + (long long)tt * K4 + laneHi * 8;
  const u16* bBase = w4 + (long long)laneLo * K4 + laneHi * 8;

  v8f acc = { 0.f, 0.f, 0.f, 0.f, 0.f, 0.f, 0.f, 0.f };
#pragma unroll
  for (int kk = 0; kk < K4 / 32; ++kk) {
    Frag fa, fb;
    fa.q[0] = *(const u32x4*)(aBase + kk * 32);
    fa.q[1] = *(const u32x4*)(aBase + kk * 32 + 16);
    fb.q[0] = *(const u32x4*)(bBase + kk * 32);
    fb.q[1] = *(const u32x4*)(bBase + kk * 32 + 16);
    acc = __builtin_amdgcn_wmma_f32_16x16x32_bf16(false, fa.v, false, fb.v,
                                                  (short)0, acc, false, false);
  }
  float bb = b4[laneLo];                 // col N = pool index p
  float v[8];
#pragma unroll
  for (int rI = 0; rI < 8; ++rI) v[rI] = acc[rI] + bb;
  // max over N (pool): lanes within each 16-lane half differ only in N
#pragma unroll
  for (int m = 1; m < 16; m <<= 1) {
#pragma unroll
    for (int rI = 0; rI < 8; ++rI)
      v[rI] = fmaxf(v[rI], __shfl_xor(v[rI], m, 32));
  }
  if (laneLo == 0) {
#pragma unroll
    for (int rI = 0; rI < 8; ++rI) {
      int t = t0 + rI + laneHi * 8;
      if (t < T) a[t] = v[rI];
    }
  }
}

// first-occurrence argmax over T (jnp.argmax semantics)
__global__ __launch_bounds__(1024)
void argmax_k(const float* __restrict__ a, int T, int* __restrict__ out) {
  __shared__ float sv[1024];
  __shared__ int   si[1024];
  float best = -INFINITY; int bi = 0;
  for (int t = threadIdx.x; t < T; t += 1024) {
    float vv = a[t];
    if (vv > best) { best = vv; bi = t; }      // strict > keeps first occurrence
  }
  sv[threadIdx.x] = best; si[threadIdx.x] = bi;
  __syncthreads();
  for (int s = 512; s > 0; s >>= 1) {
    if ((int)threadIdx.x < s) {
      float ov = sv[threadIdx.x + s]; int oi = si[threadIdx.x + s];
      if (ov > sv[threadIdx.x] || (ov == sv[threadIdx.x] && oi < si[threadIdx.x])) {
        sv[threadIdx.x] = ov; si[threadIdx.x] = oi;
      }
    }
    __syncthreads();
  }
  if (threadIdx.x == 0) *out = si[0];
}

// alphas[it] = done ? alphas[it-1] : ca ; betas likewise (done = pre-iteration flag)
__global__ void commit_rows(float* __restrict__ out, int it,
                            const float* __restrict__ aS, const float* __restrict__ aE,
                            const int* __restrict__ scal, int T) {
  int t = blockIdx.x * blockDim.x + threadIdx.x;
  if (t >= T) return;
  int d = scal[4];
  float av = (d && it > 0) ? out[(long long)(it - 1) * T + t] : aS[t];
  float bv = (d && it > 0) ? out[(long long)4 * T + (long long)(it - 1) * T + t] : aE[t];
  out[(long long)it * T + t] = av;
  out[(long long)4 * T + (long long)it * T + t] = bv;
}

// freeze logic: if !done { h=ch; c=cc; start=ns; end=ne }; done |= (ns==start_old && ne==end_old)
__global__ __launch_bounds__(256)
void commit_scalar(float* __restrict__ h, float* __restrict__ c,
                   const float* __restrict__ ch, const float* __restrict__ cc,
                   int* __restrict__ scal, int computeDone) {
  __shared__ int sd;
  if (threadIdx.x == 0) sd = scal[4];
  __syncthreads();
  int d = sd;
  if (!d)
    for (int i = threadIdx.x; i < H; i += blockDim.x) { h[i] = ch[i]; c[i] = cc[i]; }
  if (threadIdx.x == 0) {
    int ns = scal[2], ne = scal[3];
    int nd = d | (computeDone && (ns == scal[0]) && (ne == scal[1]));
    if (!d) { scal[0] = ns; scal[1] = ne; }
    scal[4] = nd;
  }
}

// ---------------- host-side orchestration ----------------
extern "C" void kernel_launch(void* const* d_in, const int* in_sizes, int n_in,
                              void* d_out, int out_size, void* d_ws, size_t ws_size,
                              hipStream_t stream) {
  (void)n_in; (void)out_size; (void)ws_size;
  const float* U   = (const float*)d_in[0];
  const float* Wih = (const float*)d_in[1];
  const float* Whh = (const float*)d_in[2];
  const float* bih = (const float*)d_in[3];
  const float* bhh = (const float*)d_in[4];
  const float* fc1w[2] = { (const float*)d_in[5],  (const float*)d_in[12] };
  const float* fc2w[2] = { (const float*)d_in[6],  (const float*)d_in[13] };
  const float* fc2b[2] = { (const float*)d_in[7],  (const float*)d_in[14] };
  const float* fc3w[2] = { (const float*)d_in[8],  (const float*)d_in[15] };
  const float* fc3b[2] = { (const float*)d_in[9],  (const float*)d_in[16] };
  const float* fc4w[2] = { (const float*)d_in[10], (const float*)d_in[17] };
  const float* fc4b[2] = { (const float*)d_in[11], (const float*)d_in[18] };

  const int T = in_sizes[0] / 400;
  const int tTiles = (T + 15) / 16;
  const int gBlocks = (tTiles + MT - 1) / MT;

  // bump allocator over workspace (total ~52 MB for T=16384)
  char* wp = (char*)d_ws;
  auto carve = [&](size_t bytes) -> void* {
    void* r = (void*)wp;
    wp += (bytes + 255) & ~(size_t)255;
    return r;
  };
  u16* Ubf = (u16*)carve((size_t)T * K1 * sizeof(u16));
  u16 *w2a[2], *w3b[2], *w4b[2], *mm[2];
  float *rvec[2], *rw2v[2], *avec[2];
  for (int hd = 0; hd < 2; ++hd) {
    w2a[hd]  = (u16*)carve((size_t)NOUT * K1 * sizeof(u16));
    w3b[hd]  = (u16*)carve((size_t)NOUT * K2 * sizeof(u16));
    w4b[hd]  = (u16*)carve((size_t)16 * K4 * sizeof(u16));
    mm[hd]   = (u16*)carve((size_t)T * K4 * sizeof(u16));
    rvec[hd] = (float*)carve(H * sizeof(float));
    rw2v[hd] = (float*)carve(NOUT * sizeof(float));
    avec[hd] = (float*)carve((size_t)T * sizeof(float));
  }
  float* hv  = (float*)carve(H * sizeof(float));
  float* cv  = (float*)carve(H * sizeof(float));
  float* chv = (float*)carve(H * sizeof(float));
  float* ccv = (float*)carve(H * sizeof(float));
  int*   scal = (int*)carve(8 * sizeof(int));

  // ---- preprocessing (deterministic, every launch) ----
  {
    long long n = (long long)T * K1;
    convW<<<(unsigned)((n + 255) / 256), 256, 0, stream>>>(U, 400, 400, Ubf, K1, n);
  }
  for (int hd = 0; hd < 2; ++hd) {
    long long n1 = (long long)NOUT * K1;
    convW<<<(unsigned)((n1 + 255) / 256), 256, 0, stream>>>(fc2w[hd], 600, 400, w2a[hd], K1, n1);
    long long n2 = (long long)NOUT * K2;
    convW<<<(unsigned)((n2 + 255) / 256), 256, 0, stream>>>(fc3w[hd], 200, 200, w3b[hd], K2, n2);
    convW4<<<(16 * K4 + 255) / 256, 256, 0, stream>>>(fc4w[hd], w4b[hd]);
    (void)hipMemsetAsync(mm[hd], 0, (size_t)T * K4 * sizeof(u16), stream); // zero pad cols
  }
  init_state<<<1, 256, 0, stream>>>(hv, cv, scal);

  float* outF = (float*)d_out;
  for (int it = 0; it < 4; ++it) {
    lstm_step_k<<<1, 800, 0, stream>>>(U, Wih, Whh, bih, bhh, scal, hv, cv, chv, ccv);
    for (int hd = 0; hd < 2; ++hd) {
      hmn_r<<<1, 256, 0, stream>>>(fc1w[hd], chv, U, scal, rvec[hd]);
      rw2_k<<<(NOUT + 255) / 256, 256, 0, stream>>>(rvec[hd], fc2w[hd], fc2b[hd], rw2v[hd]);
      // fc2 (K=416) + fused r-vector + bias + maxpool -> m1 (bf16 cols 0..199 of mm)
      gemm_pool<<<gBlocks, NJ * 32, 0, stream>>>(Ubf, K1, w2a[hd], rw2v[hd],
                                                 mm[hd], K4, 0, K1 / 32, T);
      // fc3 (K=224) + bias + maxpool -> m2 (bf16 cols 224..423 of mm)
      gemm_pool<<<gBlocks, NJ * 32, 0, stream>>>(mm[hd], K4, w3b[hd], fc3b[hd],
                                                 mm[hd], K4, K2, K2 / 32, T);
      // fc4 (K=448, N=16) + bias + maxpool -> a[t]
      fc4_max<<<tTiles, 32, 0, stream>>>(mm[hd], w4b[hd], fc4b[hd], avec[hd], T);
      argmax_k<<<1, 1024, 0, stream>>>(avec[hd], T, &scal[2 + hd]);
    }
    commit_rows<<<(T + 255) / 256, 256, 0, stream>>>(outF, it, avec[0], avec[1], scal, T);
    commit_scalar<<<1, 256, 0, stream>>>(hv, cv, chv, ccv, scal, it > 0 ? 1 : 0);
  }
}
